// ContinuousThoughtMachine_3642132267103
// MI455X (gfx1250) — matrix operand order repaired and checked by reading, not compiled, not run on hardware
//
#include <hip/hip_runtime.h>
#include <hip/hip_bf16.h>
#include <math.h>

// ---------------- model constants (from reference) ----------------
constexpr int B     = 32;
constexpr int C     = 512;
constexpr int S     = 256;     // Hf*Wf kv tokens
constexpr int D     = 2048;    // d_model
constexpr int DIN   = 512;     // attention embed dim
constexpr int HEADS = 8;
constexpr int DH    = DIN / HEADS;  // 64
constexpr int T     = 50;
constexpr int M     = 25;      // memory length
constexpr int HH    = 16;      // memory hidden dims
constexpr int NSA   = 512;
constexpr int NSO   = 512;
constexpr int OUT   = 1000;

typedef _Float16 v16h __attribute__((ext_vector_type(16)));
typedef _Float16 v8h  __attribute__((ext_vector_type(8)));
typedef float    v8f  __attribute__((ext_vector_type(8)));

// =====================================================================
// One-time weight transpose + f16 quantize:  wt[n*K + k] = (f16)w[k*N + n]
// =====================================================================
__global__ void wtrans_f16(const float* __restrict__ w, _Float16* __restrict__ wt,
                           int K, int N)
{
    const int idx = blockIdx.x * 256 + threadIdx.x;
    if (idx >= K * N) return;
    const int k = idx / N, n = idx % N;
    wt[(size_t)n * K + k] = (_Float16)w[idx];
}

// =====================================================================
// One-time feats transpose: fh[(bi*S+si)*C + c] = (f16)x[(bi*C + c)*S + si]
// =====================================================================
__global__ void xtrans_f16(const float* __restrict__ x, _Float16* __restrict__ fh)
{
    const int idx = blockIdx.x * 256 + threadIdx.x;
    if (idx >= B * S * C) return;
    const int c = idx % C;
    const int row = idx / C;
    const int bi = row / S, si = row % S;
    fh[idx] = (_Float16)x[((size_t)bi * C + c) * S + si];
}

// =====================================================================
// Pure-f16-fragment WMMA GEMM:  C = Ah(Mrows x K) @ WT^T + bias
//   Ah : f16 row-major (M x K);  WT : f16 (N x K)  (pre-transposed)
//   one wave computes (TM*16) x (TN*16) of C; all fragment data arrives
//   via global_load_b128, zero VALU in the K loop.
//   N tail: clamp B row (an OOB column of B only feeds an unstored
//   output column).  K % 32 == 0; Mrows % (16*TM) == 0.
//   Cf (f32) and Ch (f16 mirror) outputs are each optional.
// =====================================================================
template <int TM, int TN>
__global__ __launch_bounds__(32)
void gemm_wmma_h(const _Float16* __restrict__ Ah, const _Float16* __restrict__ WT,
                 const float* __restrict__ bias, float* __restrict__ Cf,
                 _Float16* __restrict__ Ch, int Mrows, int N, int K)
{
    const int lane = threadIdx.x & 31;
    const int h    = lane >> 4;      // half-wave: 0 or 1
    const int l    = lane & 15;
    const int tn0  = blockIdx.x * TN;
    const int tm0  = blockIdx.y * TM;

    int colv[TN], colc[TN];
#pragma unroll
    for (int j = 0; j < TN; ++j) {
        colv[j] = (tn0 + j) * 16 + l;
        colc[j] = (colv[j] < N) ? colv[j] : (N - 1);   // clamp, no branch
    }
    int rowv[TM];
#pragma unroll
    for (int i = 0; i < TM; ++i) rowv[i] = (tm0 + i) * 16 + l;

    v8f acc[TM][TN];
    const v8f vzero = {};
#pragma unroll
    for (int i = 0; i < TM; ++i)
#pragma unroll
        for (int j = 0; j < TN; ++j) acc[i][j] = vzero;

    for (int k = 0; k < K; k += 32) {
        // ---- B fragments: lane holds column colc[j],
        //      elements 0..15 -> K = k + h*16 + e   == 16 consecutive halves
        v16h bf[TN];
#pragma unroll
        for (int j = 0; j < TN; ++j) {
            const v8h* bp = (const v8h*)(WT + (size_t)colc[j] * K + k + h * 16);
            const v8h b0 = bp[0], b1 = bp[1];
            bf[j] = __builtin_shufflevector(b0, b1,
                        0, 1, 2, 3, 4, 5, 6, 7, 8, 9, 10, 11, 12, 13, 14, 15);
        }
        // ---- A fragments: elements 0..15 -> K = k + h*16 + e
#pragma unroll
        for (int i = 0; i < TM; ++i) {
            const v8h a0 = *(const v8h*)(Ah + (size_t)rowv[i] * K + k + h * 16);
            const v8h a1 = *(const v8h*)(Ah + (size_t)rowv[i] * K + k + h * 16 + 8);
            const v16h af = __builtin_shufflevector(a0, a1,
                        0, 1, 2, 3, 4, 5, 6, 7, 8, 9, 10, 11, 12, 13, 14, 15);
#pragma unroll
            for (int j = 0; j < TN; ++j)
                acc[i][j] = __builtin_amdgcn_wmma_f32_16x16x32_f16(
                    false, af, false, bf[j], (short)0, acc[i][j], false, false);
        }
    }

    // ---- epilogue: C/D layout: VGPR r, lanes 0-15 -> M=r, lanes 16-31 -> M=8+r
#pragma unroll
    for (int j = 0; j < TN; ++j) {
        if (colv[j] < N) {
            const float bc = bias[colv[j]];
#pragma unroll
            for (int i = 0; i < TM; ++i) {
#pragma unroll
                for (int r = 0; r < 8; ++r) {
                    const int orow = (tm0 + i) * 16 + h * 8 + r;
                    const float val = acc[i][j][r] + bc;
                    if (Cf) Cf[(size_t)orow * N + colv[j]] = val;
                    if (Ch) Ch[(size_t)orow * N + colv[j]] = (_Float16)val;
                }
            }
        }
    }
}

// =====================================================================
// Row LayerNorm (in place) + optional f16 mirror.  One block per row.
// =====================================================================
__global__ __launch_bounds__(256)
void ln_rows(float* __restrict__ X, const float* __restrict__ g,
             const float* __restrict__ bt, int n, _Float16* __restrict__ Xh)
{
    const int row = blockIdx.x, tid = threadIdx.x;
    float* xr = X + (size_t)row * n;
    __shared__ float red[256];

    float s = 0.f;
    for (int i = tid; i < n; i += 256) s += xr[i];
    red[tid] = s; __syncthreads();
    for (int o = 128; o > 0; o >>= 1) { if (tid < o) red[tid] += red[tid + o]; __syncthreads(); }
    const float mu = red[0] / n; __syncthreads();

    float v = 0.f;
    for (int i = tid; i < n; i += 256) { float d2 = xr[i] - mu; v += d2 * d2; }
    red[tid] = v; __syncthreads();
    for (int o = 128; o > 0; o >>= 1) { if (tid < o) red[tid] += red[tid + o]; __syncthreads(); }
    const float inv = rsqrtf(red[0] / n + 1e-5f);

    for (int i = tid; i < n; i += 256) {
        const float val = (xr[i] - mu) * inv * g[i] + bt[i];
        xr[i] = val;
        if (Xh) Xh[(size_t)row * n + i] = (_Float16)val;
    }
}

// =====================================================================
// GLU over 2D=4096 -> D=2048, then LayerNorm(D).  One block per batch row.
// =====================================================================
__global__ __launch_bounds__(256)
void glu_ln(const float* __restrict__ raw, const float* __restrict__ g,
            const float* __restrict__ bt, float* __restrict__ out)
{
    const int row = blockIdx.x, tid = threadIdx.x;
    const float* rr = raw + (size_t)row * (2 * D);
    __shared__ float buf[D];
    __shared__ float red[256];

    float s = 0.f;
    for (int i = tid; i < D; i += 256) {
        const float a  = rr[i];
        const float b  = rr[D + i];
        const float gl = a / (1.0f + expf(-b));
        buf[i] = gl;
        s += gl;
    }
    red[tid] = s; __syncthreads();
    for (int o = 128; o > 0; o >>= 1) { if (tid < o) red[tid] += red[tid + o]; __syncthreads(); }
    const float mu = red[0] / D; __syncthreads();

    float v = 0.f;
    for (int i = tid; i < D; i += 256) { float d2 = buf[i] - mu; v += d2 * d2; }
    red[tid] = v; __syncthreads();
    for (int o = 128; o > 0; o >>= 1) { if (tid < o) red[tid] += red[tid + o]; __syncthreads(); }
    const float inv = rsqrtf(red[0] / D + 1e-5f);

    for (int i = tid; i < D; i += 256)
        out[(size_t)row * D + i] = (buf[i] - mu) * inv * g[i] + bt[i];
}

// =====================================================================
// Single-query multi-head attention.  One block per (b, head).
// Output written as f16 (feeds the Wo GEMM only).
// =====================================================================
__global__ __launch_bounds__(256)
void attn_kernel(const float* __restrict__ qh, const float* __restrict__ kh,
                 const float* __restrict__ vh, _Float16* __restrict__ oh)
{
    const int bh = blockIdx.x;
    const int b  = bh / HEADS, hd = bh % HEADS;
    const int tid = threadIdx.x;
    __shared__ float qs[DH];
    __shared__ float sc[S];
    __shared__ float red[256];

    if (tid < DH) qs[tid] = qh[(size_t)b * DIN + hd * DH + tid];
    __syncthreads();

    {   // scores: one kv token per thread
        const float* kr = kh + ((size_t)(b * S + tid)) * DIN + hd * DH;
        float dot = 0.f;
#pragma unroll 8
        for (int d2 = 0; d2 < DH; ++d2) dot += qs[d2] * kr[d2];
        sc[tid] = dot * 0.125f;          // 1/sqrt(64)
    }
    __syncthreads();

    red[tid] = sc[tid]; __syncthreads();
    for (int o2 = 128; o2 > 0; o2 >>= 1) { if (tid < o2) red[tid] = fmaxf(red[tid], red[tid + o2]); __syncthreads(); }
    const float mx = red[0]; __syncthreads();

    const float e = expf(sc[tid] - mx);
    sc[tid] = e;
    red[tid] = e; __syncthreads();
    for (int o2 = 128; o2 > 0; o2 >>= 1) { if (tid < o2) red[tid] += red[tid + o2]; __syncthreads(); }
    const float inv = 1.0f / red[0];

    if (tid < DH) {
        float acc = 0.f;
        for (int s2 = 0; s2 < S; ++s2)
            acc += sc[s2] * vh[((size_t)(b * S + s2)) * DIN + hd * DH + tid];
        oh[(size_t)b * DIN + hd * DH + tid] = (_Float16)(acc * inv);
    }
}

// =====================================================================
// Concat pre_h = f16([attn_out (B,DIN) | act (B,D)])
// =====================================================================
__global__ void concat_pre(const float* __restrict__ ao, const float* __restrict__ act,
                           _Float16* __restrict__ pre)
{
    const int idx = blockIdx.x * blockDim.x + threadIdx.x;
    if (idx >= B * (DIN + D)) return;
    const int b = idx / (DIN + D), k = idx % (DIN + D);
    const float v = (k < DIN) ? ao[(size_t)b * DIN + k] : act[(size_t)b * D + (k - DIN)];
    pre[idx] = (_Float16)v;
}

// =====================================================================
// Per-neuron NLM over the memory ring buffer.  One thread per (b,d).
// Logical memory index m maps to physical slot (t+1+m) % M.
// =====================================================================
__global__ __launch_bounds__(256)
void nlm_kernel(const float* __restrict__ state, float* __restrict__ trace,
                float* __restrict__ act,
                const float* __restrict__ w1, const float* __restrict__ b1,
                const float* __restrict__ w2, const float* __restrict__ b2,
                int t)
{
    const int idx = blockIdx.x * 256 + threadIdx.x;   // b*D + d
    if (idx >= B * D) return;
    const int d = idx % D;
    const float sn = state[idx];
    const size_t tb = (size_t)idx * M;
    trace[tb + (t % M)] = sn;

    float tr[M];
#pragma unroll
    for (int m = 0; m < M - 1; ++m)
        tr[m] = trace[tb + ((t + 1 + m) % M)];
    tr[M - 1] = sn;

    float hp[2 * HH];
#pragma unroll
    for (int j = 0; j < 2 * HH; ++j) hp[j] = b1[(size_t)d * (2 * HH) + j];
    for (int m = 0; m < M; ++m) {
        const float tv = tr[m];
        const float* wrow = w1 + (size_t)m * (2 * HH) * D + d;
#pragma unroll
        for (int j = 0; j < 2 * HH; ++j) hp[j] += tv * wrow[(size_t)j * D];
    }
    float hv[HH];
#pragma unroll
    for (int j = 0; j < HH; ++j)
        hv[j] = hp[j] / (1.0f + expf(-hp[HH + j]));      // GLU

    float o0 = b2[(size_t)d * 2 + 0];
    float o1 = b2[(size_t)d * 2 + 1];
#pragma unroll
    for (int j = 0; j < HH; ++j) {
        o0 += hv[j] * w2[(size_t)(j * 2 + 0) * D + d];
        o1 += hv[j] * w2[(size_t)(j * 2 + 1) * D + d];
    }
    act[idx] = o0 / (1.0f + expf(-o1));                  // GLU -> [...,0]
}

// =====================================================================
// Synchronisation update:  a = r*a + act[il]*act[ir]; b = r*b + 1;
// sync = a/sqrt(b) written as f16 (feeds a GEMM); optional f32 tail copy.
// =====================================================================
__global__ void sync_update(const float* __restrict__ act, const float* __restrict__ decay,
                            const int* __restrict__ il, const int* __restrict__ ir,
                            float* __restrict__ av, float* __restrict__ bv,
                            _Float16* __restrict__ sync_h, int Nsync,
                            float* __restrict__ out_tail)
{
    const int idx = blockIdx.x * blockDim.x + threadIdx.x;
    if (idx >= B * Nsync) return;
    const int b = idx / Nsync, j = idx % Nsync;
    const float r  = expf(-fminf(fmaxf(decay[j], 0.0f), 15.0f));
    const float p  = act[(size_t)b * D + il[j]] * act[(size_t)b * D + ir[j]];
    const float a2 = r * av[idx] + p;
    const float b2 = r * bv[idx] + 1.0f;
    av[idx] = a2; bv[idx] = b2;
    const float s = a2 * rsqrtf(b2);
    sync_h[idx] = (_Float16)s;
    if (out_tail) out_tail[idx] = s;
}

// =====================================================================
// Output post-processing: logits -> predictions[:, :, t], entropy ->
// certainties[:, :, t].  One block per batch row.
// =====================================================================
__global__ __launch_bounds__(256)
void pred_post(const float* __restrict__ pred, float* __restrict__ outp, int t)
{
    const int b = blockIdx.x, tid = threadIdx.x;
    const float* pr = pred + (size_t)b * OUT;
    __shared__ float red[256];

    float mx = -3.4e38f;
    for (int i = tid; i < OUT; i += 256) mx = fmaxf(mx, pr[i]);
    red[tid] = mx; __syncthreads();
    for (int o = 128; o > 0; o >>= 1) { if (tid < o) red[tid] = fmaxf(red[tid], red[tid + o]); __syncthreads(); }
    mx = red[0]; __syncthreads();

    float se = 0.f;
    for (int i = tid; i < OUT; i += 256) se += expf(pr[i] - mx);
    red[tid] = se; __syncthreads();
    for (int o = 128; o > 0; o >>= 1) { if (tid < o) red[tid] += red[tid + o]; __syncthreads(); }
    const float logZ = mx + logf(red[0]); __syncthreads();

    float ent = 0.f;
    for (int i = tid; i < OUT; i += 256) { const float lp = pr[i] - logZ; ent += expf(lp) * lp; }
    red[tid] = ent; __syncthreads();
    for (int o = 128; o > 0; o >>= 1) { if (tid < o) red[tid] += red[tid + o]; __syncthreads(); }
    const float ne = -red[0] / logf((float)OUT);

    float* po = outp + (size_t)b * OUT * T;
    for (int i = tid; i < OUT; i += 256) po[(size_t)i * T + t] = pr[i];
    if (tid == 0) {
        float* ct = outp + (size_t)B * OUT * T + (size_t)b * 2 * T;
        ct[t] = ne;
        ct[T + t] = 1.0f - ne;
    }
}

// =====================================================================
// Initial state: trace <- start_trace, act <- start_act, a/b counters.
// =====================================================================
__global__ void init_state(float* __restrict__ trace, float* __restrict__ act,
                           float* __restrict__ a_a, float* __restrict__ b_a,
                           float* __restrict__ a_o, float* __restrict__ b_o,
                           const float* __restrict__ start_trace,
                           const float* __restrict__ start_act,
                           const int* __restrict__ ilo, const int* __restrict__ iro)
{
    const int idx = blockIdx.x * blockDim.x + threadIdx.x;
    if (idx < B * D * M) trace[idx] = start_trace[idx % (D * M)];
    if (idx < B * D)     act[idx]   = start_act[idx % D];
    if (idx < B * NSA)   { a_a[idx] = 0.0f; b_a[idx] = 0.0f; }
    if (idx < B * NSO) {
        const int j = idx % NSO;
        a_o[idx] = start_act[ilo[j]] * start_act[iro[j]];
        b_o[idx] = 1.0f;
    }
}

// =====================================================================
// Host launcher
// =====================================================================
extern "C" void kernel_launch(void* const* d_in, const int* in_sizes, int n_in,
                              void* d_out, int out_size, void* d_ws, size_t ws_size,
                              hipStream_t stream)
{
    (void)in_sizes; (void)n_in; (void)out_size; (void)ws_size;
    const float* x        = (const float*)d_in[0];
    const float* kv_w     = (const float*)d_in[1];
    const float* kv_b     = (const float*)d_in[2];
    const float* ln_kv_g  = (const float*)d_in[3];
    const float* ln_kv_b  = (const float*)d_in[4];
    const float* q_w      = (const float*)d_in[5];
    const float* q_b      = (const float*)d_in[6];
    const float* Wq       = (const float*)d_in[7];
    const float* bq       = (const float*)d_in[8];
    const float* Wk       = (const float*)d_in[9];
    const float* bk       = (const float*)d_in[10];
    const float* Wv       = (const float*)d_in[11];
    const float* bv       = (const float*)d_in[12];
    const float* Wo       = (const float*)d_in[13];
    const float* bo       = (const float*)d_in[14];
    const float* syn_w    = (const float*)d_in[15];
    const float* syn_b    = (const float*)d_in[16];
    const float* ln_syn_g = (const float*)d_in[17];
    const float* ln_syn_b = (const float*)d_in[18];
    const float* nlm_w1   = (const float*)d_in[19];
    const float* nlm_b1   = (const float*)d_in[20];
    const float* nlm_w2   = (const float*)d_in[21];
    const float* nlm_b2   = (const float*)d_in[22];
    const float* out_w    = (const float*)d_in[23];
    const float* out_b    = (const float*)d_in[24];
    const float* decay_a  = (const float*)d_in[25];
    const float* decay_o  = (const float*)d_in[26];
    const float* st_trace = (const float*)d_in[27];
    const float* st_act   = (const float*)d_in[28];
    const int*   idx_la   = (const int*)d_in[29];
    const int*   idx_ra   = (const int*)d_in[30];
    const int*   idx_lo   = (const int*)d_in[31];
    const int*   idx_ro   = (const int*)d_in[32];

    float* outf = (float*)d_out;

    // -------- workspace carve-out: f32 region, then f16 region --------
    float* wsf = (float*)d_ws;
    size_t off = 0;
    auto carve = [&](size_t n) { float* p = wsf + off; off += n; return p; };
    float* kv      = carve((size_t)B * S * DIN);   // f32 kv (LN in place)
    float* khb     = carve((size_t)B * S * DIN);
    float* vhb     = carve((size_t)B * S * DIN);
    float* trace   = carve((size_t)B * D * M);
    float* act     = carve((size_t)B * D);
    float* a_a     = carve((size_t)B * NSA);
    float* b_a     = carve((size_t)B * NSA);
    float* a_o     = carve((size_t)B * NSO);
    float* b_o     = carve((size_t)B * NSO);
    float* qhbuf   = carve((size_t)B * DIN);
    float* attnout = carve((size_t)B * DIN);
    float* synraw  = carve((size_t)B * 2 * D);
    float* state   = carve((size_t)B * D);
    float* predb   = carve((size_t)B * OUT);

    _Float16* wsh = (_Float16*)(wsf + off);
    size_t hoff = 0;
    auto carveh = [&](size_t n) { _Float16* p = wsh + hoff; hoff += n; return p; };
    _Float16* featsh   = carveh((size_t)B * S * C);
    _Float16* kv_h     = carveh((size_t)B * S * DIN);
    _Float16* kv_wT    = carveh((size_t)C * DIN);
    _Float16* q_wT     = carveh((size_t)NSA * DIN);
    _Float16* WqT      = carveh((size_t)DIN * DIN);
    _Float16* WkT      = carveh((size_t)DIN * DIN);
    _Float16* WvT      = carveh((size_t)DIN * DIN);
    _Float16* WoT      = carveh((size_t)DIN * DIN);
    _Float16* syn_wT   = carveh((size_t)(DIN + D) * 2 * D);
    _Float16* out_wT   = carveh((size_t)NSO * OUT);
    _Float16* sync_a_h = carveh((size_t)B * NSA);
    _Float16* sync_o_h = carveh((size_t)B * NSO);
    _Float16* qbuf_h   = carveh((size_t)B * DIN);
    _Float16* obuf_h   = carveh((size_t)B * DIN);
    _Float16* pre_h    = carveh((size_t)B * (DIN + D));

    const dim3 wave(32);

    // -------- one-time: transpose/quantize weights + feats --------
    xtrans_f16<<<(B * S * C + 255) / 256, 256, 0, stream>>>(x, featsh);
    wtrans_f16<<<(C * DIN + 255) / 256, 256, 0, stream>>>(kv_w, kv_wT, C, DIN);
    wtrans_f16<<<(NSA * DIN + 255) / 256, 256, 0, stream>>>(q_w, q_wT, NSA, DIN);
    wtrans_f16<<<(DIN * DIN + 255) / 256, 256, 0, stream>>>(Wq, WqT, DIN, DIN);
    wtrans_f16<<<(DIN * DIN + 255) / 256, 256, 0, stream>>>(Wk, WkT, DIN, DIN);
    wtrans_f16<<<(DIN * DIN + 255) / 256, 256, 0, stream>>>(Wv, WvT, DIN, DIN);
    wtrans_f16<<<(DIN * DIN + 255) / 256, 256, 0, stream>>>(Wo, WoT, DIN, DIN);
    wtrans_f16<<<((DIN + D) * 2 * D + 255) / 256, 256, 0, stream>>>(
        syn_w, syn_wT, DIN + D, 2 * D);
    wtrans_f16<<<(NSO * OUT + 255) / 256, 256, 0, stream>>>(out_w, out_wT, NSO, OUT);

    // -------- precompute: kv = LN(feats @ kv_w + kv_b); K = kv@Wk; V = kv@Wv
    gemm_wmma_h<4, 4><<<dim3(DIN / 64, (B * S) / 64), wave, 0, stream>>>(
        featsh, kv_wT, kv_b, kv, nullptr, B * S, DIN, C);
    ln_rows<<<B * S, 256, 0, stream>>>(kv, ln_kv_g, ln_kv_b, DIN, kv_h);
    gemm_wmma_h<4, 4><<<dim3(DIN / 64, (B * S) / 64), wave, 0, stream>>>(
        kv_h, WkT, bk, khb, nullptr, B * S, DIN, DIN);
    gemm_wmma_h<4, 4><<<dim3(DIN / 64, (B * S) / 64), wave, 0, stream>>>(
        kv_h, WvT, bv, vhb, nullptr, B * S, DIN, DIN);

    init_state<<<(B * D * M + 255) / 256, 256, 0, stream>>>(
        trace, act, a_a, b_a, a_o, b_o, st_trace, st_act, idx_lo, idx_ro);

    float* sync_tail = outf + (size_t)B * OUT * T + (size_t)B * 2 * T;

    // -------- T internal ticks --------
    for (int t = 0; t < T; ++t) {
        sync_update<<<(B * NSA + 255) / 256, 256, 0, stream>>>(
            act, decay_a, idx_la, idx_ra, a_a, b_a, sync_a_h, NSA, nullptr);

        gemm_wmma_h<2, 4><<<dim3(DIN / 64, 1), wave, 0, stream>>>(
            sync_a_h, q_wT, q_b, nullptr, qbuf_h, B, DIN, NSA);
        gemm_wmma_h<2, 4><<<dim3(DIN / 64, 1), wave, 0, stream>>>(
            qbuf_h, WqT, bq, qhbuf, nullptr, B, DIN, DIN);

        attn_kernel<<<B * HEADS, 256, 0, stream>>>(qhbuf, khb, vhb, obuf_h);

        gemm_wmma_h<2, 4><<<dim3(DIN / 64, 1), wave, 0, stream>>>(
            obuf_h, WoT, bo, attnout, nullptr, B, DIN, DIN);

        concat_pre<<<(B * (DIN + D) + 255) / 256, 256, 0, stream>>>(attnout, act, pre_h);

        gemm_wmma_h<2, 4><<<dim3((2 * D) / 64, 1), wave, 0, stream>>>(
            pre_h, syn_wT, syn_b, synraw, nullptr, B, 2 * D, DIN + D);

        glu_ln<<<B, 256, 0, stream>>>(synraw, ln_syn_g, ln_syn_b, state);

        nlm_kernel<<<(B * D + 255) / 256, 256, 0, stream>>>(
            state, trace, act, nlm_w1, nlm_b1, nlm_w2, nlm_b2, t);

        sync_update<<<(B * NSO + 255) / 256, 256, 0, stream>>>(
            act, decay_o, idx_lo, idx_ro, a_o, b_o, sync_o_h, NSO,
            (t == T - 1) ? sync_tail : nullptr);

        gemm_wmma_h<2, 4><<<dim3((OUT + 63) / 64, 1), wave, 0, stream>>>(
            sync_o_h, out_wT, out_b, predb, nullptr, B, OUT, NSO);

        pred_post<<<B, 256, 0, stream>>>(predb, outf, t);
    }
}
